// UNetInnerProd_60971355734445
// MI455X (gfx1250) — compile-verified, matrix operand order checked by reading, not run on hardware
//
#include <hip/hip_runtime.h>
#include <math.h>

// ---------------- problem constants (from reference) ----------------
#define WW    96
#define HH    72
#define NPIX  6912          // 72*96
#define CF    16            // feature channels
#define MT    432           // NPIX / 16 : number of 16-wide tiles
#define FXI   48.0f
#define FYI   48.0f
#define CXI   48.0f
#define CYI   36.0f
#define EPSN  1e-8f
#define LOG2E 1.4426950408889634f

// j-chunking: each wave owns one i-tile and one chunk of j-tiles
#define NCHUNK 8
#define JCHUNK (MT / NCHUNK)           // 54
#define MAIN_BLOCKS (MT * NCHUNK / 8)  // 8 waves/block -> 432 blocks

typedef __attribute__((ext_vector_type(16))) _Float16 v16h;
typedef __attribute__((ext_vector_type(8)))  float    v8f;
typedef __attribute__((ext_vector_type(4)))  float    v4f;
typedef __attribute__((ext_vector_type(2)))  float    v2f;

// ---------------------------------------------------------------------------
// Prep: per pixel, back-project with depth, apply pose / pose*noise, L2-
// normalize features, and pack operands directly in WMMA per-lane VGPR order.
//
// f16 A/B 16x32 lane layout (ISA 7.12.2): lane<16 holds M=lane, K = 0..7 then
// 16..23; lane>=16 holds M=lane-16, K = 8..15 then 24..31.  16 f16 stored
// contiguously per (tile,lane) in that order (K>=16 entries = zero pad).
//
// f32 A/B 16x4 lane layout: lane<16 -> {K0,K1}, lane>=16 -> {K2,K3}.  4 f32
// per (tile,lane): [pair for wmma #1, pair for wmma #2] covering padded K=8.
//
// A-side geometry rows are pre-scaled by log2(e) so the WMMA accumulator
// directly yields log2(kern); main loop needs only v_exp per value.
// ---------------------------------------------------------------------------
__global__ void ip_prep_kernel(const float* __restrict__ f1,
                               const float* __restrict__ f2,
                               const float* __restrict__ d1,
                               const float* __restrict__ d2,
                               const float* __restrict__ pose,
                               const float* __restrict__ noise,
                               _Float16* __restrict__ Ag16,
                               _Float16* __restrict__ Bg16,
                               float* __restrict__ AgF,
                               float* __restrict__ BtF,
                               float* __restrict__ BnF,
                               float* __restrict__ norm1,
                               float* __restrict__ norm2) {
  int m = blockIdx.x * blockDim.x + threadIdx.x;
  if (m >= NPIX) return;
  int u = m % WW;
  int v = m / WW;

  // yz1 grid = inv(K) @ [u,v,1] for K=[[cx,fx,0],[cy,0,fy],[1,0,0]]
  float p0 = 1.0f;
  float p1 = ((float)u - CXI) / FXI;
  float p2 = ((float)v - CYI) / FYI;

  float dd1 = d1[m];
  float x1 = p0 * dd1, y1 = p1 * dd1, z1 = p2 * dd1;
  float n1 = x1 * x1 + y1 * y1 + z1 * z1;

  float dd2 = d2[m];
  float q0 = p0 * dd2, q1 = p1 * dd2, q2 = p2 * dd2;

  float P[16];
#pragma unroll
  for (int k = 0; k < 16; ++k) P[k] = pose[k];

  // xyz2 transformed by pose1_2 (homogeneous w = 1)
  float xt[3], xn[3];
#pragma unroll
  for (int r = 0; r < 3; ++r)
    xt[r] = P[r * 4 + 0] * q0 + P[r * 4 + 1] * q1 + P[r * 4 + 2] * q2 + P[r * 4 + 3];

  // pose_noisy = pose1_2 @ pose_noise (rows 0..2 suffice)
#pragma unroll
  for (int r = 0; r < 3; ++r) {
    float c0 = 0.f, c1 = 0.f, c2 = 0.f, c3 = 0.f;
#pragma unroll
    for (int k = 0; k < 4; ++k) {
      float pr = P[r * 4 + k];
      c0 += pr * noise[k * 4 + 0];
      c1 += pr * noise[k * 4 + 1];
      c2 += pr * noise[k * 4 + 2];
      c3 += pr * noise[k * 4 + 3];
    }
    xn[r] = c0 * q0 + c1 * q1 + c2 * q2 + c3;
  }

  float n2t = xt[0] * xt[0] + xt[1] * xt[1] + xt[2] * xt[2];
  float n2n = xn[0] * xn[0] + xn[1] * xn[1] + xn[2] * xn[2];

  int tile = m >> 4;
  int M = m & 15;

  // --- geometry packs (A rows scaled by LOG2E):
  //     A row: LOG2E * [x,y,z,-n1/2, 1, 0,0,0]
  //     B row:         [x,y,z,  1, -n2/2, 0,0,0]
  {
    float* a0 = AgF + (size_t)(tile * 32 + M) * 4;       // K0,K1,K4,K5
    a0[0] = LOG2E * x1; a0[1] = LOG2E * y1; a0[2] = LOG2E; a0[3] = 0.0f;
    float* a1 = AgF + (size_t)(tile * 32 + M + 16) * 4;  // K2,K3,K6,K7
    a1[0] = LOG2E * z1; a1[1] = LOG2E * -0.5f * n1; a1[2] = 0.0f; a1[3] = 0.0f;

    float* b0 = BtF + (size_t)(tile * 32 + M) * 4;
    b0[0] = xt[0]; b0[1] = xt[1]; b0[2] = -0.5f * n2t; b0[3] = 0.0f;
    float* b1 = BtF + (size_t)(tile * 32 + M + 16) * 4;
    b1[0] = xt[2]; b1[1] = 1.0f; b1[2] = 0.0f; b1[3] = 0.0f;

    float* c0 = BnF + (size_t)(tile * 32 + M) * 4;
    c0[0] = xn[0]; c0[1] = xn[1]; c0[2] = -0.5f * n2n; c0[3] = 0.0f;
    float* c1 = BnF + (size_t)(tile * 32 + M + 16) * 4;
    c1[0] = xn[2]; c1[1] = 1.0f; c1[2] = 0.0f; c1[3] = 0.0f;
  }

  // --- feature gram packs (f16, K padded 16 -> 32 with zeros)
  {
    float a[CF];
    float s = 0.f;
#pragma unroll
    for (int c = 0; c < CF; ++c) { a[c] = f1[(size_t)c * NPIX + m]; s += a[c] * a[c]; }
    float nrm = sqrtf(s);
    norm1[m] = nrm;
    float inv = 1.0f / (nrm + EPSN);
    _Float16* g0 = Ag16 + (size_t)(tile * 32 + M) * 16;       // K0..7 | K16..23(=0)
    _Float16* g1 = Ag16 + (size_t)(tile * 32 + M + 16) * 16;  // K8..15| K24..31(=0)
#pragma unroll
    for (int t = 0; t < 8; ++t) {
      g0[t] = (_Float16)(a[t] * inv);     g0[8 + t] = (_Float16)0.f;
      g1[t] = (_Float16)(a[8 + t] * inv); g1[8 + t] = (_Float16)0.f;
    }

    s = 0.f;
#pragma unroll
    for (int c = 0; c < CF; ++c) { a[c] = f2[(size_t)c * NPIX + m]; s += a[c] * a[c]; }
    nrm = sqrtf(s);
    norm2[m] = nrm;
    inv = 1.0f / (nrm + EPSN);
    _Float16* h0 = Bg16 + (size_t)(tile * 32 + M) * 16;
    _Float16* h1 = Bg16 + (size_t)(tile * 32 + M + 16) * 16;
#pragma unroll
    for (int t = 0; t < 8; ++t) {
      h0[t] = (_Float16)(a[t] * inv);     h0[8 + t] = (_Float16)0.f;
      h1[t] = (_Float16)(a[8 + t] * inv); h1[8 + t] = (_Float16)0.f;
    }
  }
}

// ---------------------------------------------------------------------------
// Tile kernel: each wave32 owns one 16-row i-tile and a 54-tile j-chunk.
// A-side operands stay in registers for the whole chunk; inner loop streams
// only B panels (~0.9 MB total, L2/WGP$-resident).  Per tile:
// 1x v_wmma_f32_16x16x32_f16 + 4x v_wmma_f32_16x16x4_f32, 16x v_exp_f32,
// and packed add/fma accumulation.
//
// NOTE on the reference's max(d2,0) clamp: d2 >= 0 mathematically; the clamp
// only guards fp rounding on near-coincident points where the exponent is
// within ~1e-6 of 0, so exp differs by a relative ~1e-6 on a few near-
// diagonal entries.  We omit it: the deviation is far below the f16 gram
// quantization and removes 32 VALU ops per tile (canonicalize + min).
// ---------------------------------------------------------------------------
__global__ void ip_tile_kernel(const _Float16* __restrict__ Ag16,
                               const _Float16* __restrict__ Bg16,
                               const float* __restrict__ AgF,
                               const float* __restrict__ BtF,
                               const float* __restrict__ BnF,
                               float* __restrict__ partials) {
  const int lane = threadIdx.x & 31;
  const int wib = threadIdx.x >> 5;
  const int wavesPerBlock = blockDim.x >> 5;
  const int wave = blockIdx.x * wavesPerBlock + wib;   // 0 .. MT*NCHUNK-1

  const int i = wave / NCHUNK;          // i-tile (0..431)
  const int chunk = wave - i * NCHUNK;  // j-chunk (0..7)
  const int j0 = chunk * JCHUNK;
  const int j1 = j0 + JCHUNK;

  // A-side operands: resident in registers for the whole chunk
  const v16h a = *(const v16h*)(Ag16 + (size_t)(i * 32 + lane) * 16);
  const v4f ga = *(const v4f*)(AgF + (size_t)(i * 32 + lane) * 4);
  const v2f a0 = {ga.x, ga.y}, a1 = {ga.z, ga.w};

  v2f acc2 = {0.0f, 0.0f};

  for (int j = j0; j < j1; ++j) {
    // gram tile (f16 WMMA, f32 accum)
    v16h b = *(const v16h*)(Bg16 + (size_t)(j * 32 + lane) * 16);
    v8f g = {};
    g = __builtin_amdgcn_wmma_f32_16x16x32_f16(false, a, false, b, (short)0, g,
                                               false, false);

    // geometry log2-exponent tiles (f32 WMMA, K=8 => 2 instructions each)
    v4f bt = *(const v4f*)(BtF + (size_t)(j * 32 + lane) * 4);
    v4f bn = *(const v4f*)(BnF + (size_t)(j * 32 + lane) * 4);
    v2f t0 = {bt.x, bt.y}, t1 = {bt.z, bt.w};
    v2f n0 = {bn.x, bn.y}, n1 = {bn.z, bn.w};

    v8f et = {};
    et = __builtin_amdgcn_wmma_f32_16x16x4_f32(false, a0, false, t0, (short)0,
                                               et, false, false);
    et = __builtin_amdgcn_wmma_f32_16x16x4_f32(false, a1, false, t1, (short)0,
                                               et, false, false);
    v8f en = {};
    en = __builtin_amdgcn_wmma_f32_16x16x4_f32(false, a0, false, n0, (short)0,
                                               en, false, false);
    en = __builtin_amdgcn_wmma_f32_16x16x4_f32(false, a1, false, n1, (short)0,
                                               en, false, false);

    // accumulator holds log2(kern) = LOG2E*(cross - n1/2 - n2/2) <= ~0.
#pragma unroll
    for (int r = 0; r < 8; r += 2) {
      v2f p  = {__builtin_amdgcn_exp2f(et[r]),
                __builtin_amdgcn_exp2f(et[r + 1])};
      v2f pn = {__builtin_amdgcn_exp2f(en[r]),
                __builtin_amdgcn_exp2f(en[r + 1])};
      v2f gg = {g[r], g[r + 1]};
      acc2 += (p - pn) * gg;   // v_pk_add (neg) + v_pk_fma
    }
  }

  float acc = acc2.x + acc2.y;

  // wave32 reduction
#pragma unroll
  for (int off = 16; off > 0; off >>= 1) acc += __shfl_down(acc, off, 32);

  __shared__ float smem[32];
  if (lane == 0) smem[wib] = acc;
  __syncthreads();
  if (threadIdx.x == 0) {
    float s = 0.f;
    for (int w = 0; w < wavesPerBlock; ++w) s += smem[w];
    partials[blockIdx.x] = s;
  }
}

// ---------------------------------------------------------------------------
// Final deterministic reduction: loss scalars + fea_norm_sum * 100.
// out = [final_loss, inner_neg, fea_norm_sum]
// ---------------------------------------------------------------------------
__global__ void ip_final_kernel(const float* __restrict__ partials, int npart,
                                const float* __restrict__ norm1,
                                const float* __restrict__ norm2,
                                float* __restrict__ out) {
  __shared__ float sh[256];
  float s = 0.f;
  for (int k = threadIdx.x; k < npart; k += blockDim.x) s += partials[k];
  sh[threadIdx.x] = s;
  __syncthreads();
  for (int st = 128; st > 0; st >>= 1) {
    if (threadIdx.x < st) sh[threadIdx.x] += sh[threadIdx.x + st];
    __syncthreads();
  }
  float total = sh[0];
  __syncthreads();

  float ns = 0.f;
  for (int k = threadIdx.x; k < NPIX; k += blockDim.x) ns += norm1[k] + norm2[k];
  sh[threadIdx.x] = ns;
  __syncthreads();
  for (int st = 128; st > 0; st >>= 1) {
    if (threadIdx.x < st) sh[threadIdx.x] += sh[threadIdx.x + st];
    __syncthreads();
  }
  if (threadIdx.x == 0) {
    // gram carried no 1/N factor and no minus sign; apply both here.
    float inner_neg = -total / (float)NPIX;
    out[0] = inner_neg;          // final_loss (sparse_mode=False)
    out[1] = inner_neg;          // inner_neg
    out[2] = sh[0] * 100.0f;     // fea_norm_sum * 100
  }
}

// ---------------------------------------------------------------------------
extern "C" void kernel_launch(void* const* d_in, const int* in_sizes, int n_in,
                              void* d_out, int out_size, void* d_ws, size_t ws_size,
                              hipStream_t stream) {
  const float* feature1 = (const float*)d_in[0];
  const float* feature2 = (const float*)d_in[1];
  const float* depth1   = (const float*)d_in[2];
  const float* depth2   = (const float*)d_in[3];
  const float* pose1_2  = (const float*)d_in[4];
  const float* posenoi  = (const float*)d_in[5];
  float* out = (float*)d_out;

  // workspace carve-up
  char* ws = (char*)d_ws;
  size_t off = 0;
  _Float16* Ag16 = (_Float16*)(ws + off); off += (size_t)MT * 32 * 16 * sizeof(_Float16); // 442368 B
  _Float16* Bg16 = (_Float16*)(ws + off); off += (size_t)MT * 32 * 16 * sizeof(_Float16);
  float* AgF   = (float*)(ws + off); off += (size_t)MT * 32 * 4 * sizeof(float);          // 221184 B
  float* BtF   = (float*)(ws + off); off += (size_t)MT * 32 * 4 * sizeof(float);
  float* BnF   = (float*)(ws + off); off += (size_t)MT * 32 * 4 * sizeof(float);
  float* norm1 = (float*)(ws + off); off += (size_t)NPIX * sizeof(float);
  float* norm2 = (float*)(ws + off); off += (size_t)NPIX * sizeof(float);
  float* partials = (float*)(ws + off); off += (size_t)MAIN_BLOCKS * sizeof(float);
  (void)ws_size; (void)in_sizes; (void)n_in; (void)out_size;

  ip_prep_kernel<<<(NPIX + 255) / 256, 256, 0, stream>>>(
      feature1, feature2, depth1, depth2, pose1_2, posenoi,
      Ag16, Bg16, AgF, BtF, BnF, norm1, norm2);

  ip_tile_kernel<<<MAIN_BLOCKS, 256, 0, stream>>>(Ag16, Bg16, AgF, BtF, BnF,
                                                  partials);

  ip_final_kernel<<<1, 256, 0, stream>>>(partials, MAIN_BLOCKS, norm1, norm2,
                                         out);
}
